// SpikingLinear_50775103373693
// MI455X (gfx1250) — matrix operand order repaired and checked by reading, compiled, not run on hardware
//
#include <hip/hip_runtime.h>

typedef __attribute__((ext_vector_type(16))) _Float16 v16h;
typedef __attribute__((ext_vector_type(8)))  _Float16 v8h;
typedef __attribute__((ext_vector_type(8)))  float    v8f;

#define IN_DIM   2048
#define OUT_DIM  2048
#define BATCH    256
#define STEPS    40
#define SBLOCK   4                 // time steps per K-pass (temporal blocking)
#define KCHUNKS  (IN_DIM / 32)     // 64 K-chunks of 32
#define OUT_TILES (OUT_DIM / 16)   // 128 tiles of 16 output cols
#define FRAG_HALVES ((size_t)OUT_TILES * KCHUNKS * 32 * 16)  // 4M halves = 8MB per plane
#define SROW     2064              // padded LDS row stride in halves (4128B, breaks bank conflicts)
#define WG_BATCH 64
#define LDS_BYTES (WG_BATCH * SROW * 2)  // 264192 B

static __device__ __forceinline__ v16h splat16(_Float16 x) {
  v16h v;
#pragma unroll
  for (int i = 0; i < 16; ++i) v[i] = x;
  return v;
}

// ---------------------------------------------------------------------------
// Prep 1: split W (f32, [OUT][IN] row-major) into hi/lo f16 planes, stored in
// the per-lane B-fragment order this code uses (contiguous k, same as A):
//   frag id = (out_tile*KCHUNKS + chunk)*32 + lane, 16 contiguous halves each.
//   lane 0-15 : N = lane,    K = chunk*32 + 0..15
//   lane 16-31: N = lane-16, K = chunk*32 + 16..31
// ---------------------------------------------------------------------------
__global__ void prep_w_kernel(const float* __restrict__ W,
                              _Float16* __restrict__ hi,
                              _Float16* __restrict__ lo) {
  int id   = blockIdx.x * blockDim.x + threadIdx.x;
  int lane = id & 31;
  int tc   = id >> 5;
  int c    = tc & (KCHUNKS - 1);
  int t    = tc >> 6;
  int n    = t * 16 + (lane & 15);
  int kh   = c * 32 + ((lane >> 4) ? 16 : 0);
  const float* src = W + (size_t)n * IN_DIM + kh;
  size_t dst = (size_t)id * 16;
#pragma unroll
  for (int j = 0; j < 16; ++j) {
    float w = src[j];
    _Float16 h = (_Float16)w;
    hi[dst + j] = h;
    lo[dst + j] = (_Float16)(w - (float)h);
  }
}

// Prep 2: spike times (i*0.5 grid) -> integer step index as f16 (exact, 0..39)
__global__ void prep_s_kernel(const float* __restrict__ inp,
                              _Float16* __restrict__ sh) {
  int id = blockIdx.x * blockDim.x + threadIdx.x;
  sh[id] = (_Float16)(inp[id] * 2.0f);
}

// ---------------------------------------------------------------------------
// Main fused SNN kernel.
// Block: 256 threads = 8 wave32. Tile: 64 batch x 32 out.
// Wave w: batch sub-tile bs = w>>1 (16 rows), out sub-tile os = w&1 (16 cols).
// Time blocked by SBLOCK=4: one K-pass produces B_i for 4 consecutive steps
// (4 f32 accumulators), amortizing W-fragment loads and LDS reads 4x.
// V, I, first-spike state lives in VGPRs across all 40 steps.
//
// NOTE (probed, negative result): writing the A-build clamp as
// min(max(1-d*d,0),1) does NOT fold into a VOP3P clamp modifier on gfx1250 —
// it emits an extra v_pk_min_num_f16 stream (+11% VALU). Keep max-only form.
// ---------------------------------------------------------------------------
__launch_bounds__(256, 1)
__global__ void snn_ttfs_kernel(const _Float16* __restrict__ fragHi,
                                const _Float16* __restrict__ fragLo,
                                const _Float16* __restrict__ sh,
                                float* __restrict__ out) {
  extern __shared__ _Float16 lds[];
  const int tid  = threadIdx.x;
  const int lane = tid & 31;
  const int w    = tid >> 5;
  const int bs   = w >> 1;                 // 0..3
  const int os   = w & 1;                  // 0..1
  const int bbase = blockIdx.y * WG_BATCH; // global batch base
  const int ot    = blockIdx.x * 2 + os;   // out tile 0..127

  // ---- stage spike-step f16 plane for our 64 batch rows into padded LDS ----
  {
    const unsigned long long* src = (const unsigned long long*)sh;  // 4 halves per u64
    for (int tix = tid; tix < WG_BATCH * (IN_DIM / 4); tix += 256) {
      int r  = tix >> 9;        // IN_DIM/4 = 512 u64 per row
      int kc = tix & 511;
      unsigned long long v = src[(((size_t)(bbase + r) * IN_DIM) >> 2) + kc];
      *(unsigned long long*)(lds + r * SROW + kc * 4) = v;
    }
  }
  __syncthreads();

  // ---- per-wave fragment addressing ----
  // A (16x32 f16): lane row M = lane&15; lane<16 -> K {0..15}, lane>=16 -> {16..31}
  const int arow = bs * 16 + (lane & 15);
  const int ag   = (lane >> 4) ? 16 : 0;
  const _Float16* aBase = lds + arow * SROW + ag;
  const v16h* fH = (const v16h*)fragHi + (size_t)ot * KCHUNKS * 32 + lane;
  const v16h* fL = (const v16h*)fragLo + (size_t)ot * KCHUNKS * 32 + lane;

  const v16h one16  = splat16((_Float16)1.0f);
  const v16h zero16 = splat16((_Float16)0.0f);

  v8f V = {}, I = {}, first = {};

  for (int base = 0; base < STEPS; base += SBLOCK) {
    v16h stepv[SBLOCK];
#pragma unroll
    for (int j = 0; j < SBLOCK; ++j) stepv[j] = splat16((_Float16)(float)(base + j));

    v8f acc[SBLOCK];
#pragma unroll
    for (int j = 0; j < SBLOCK; ++j) acc[j] = (v8f){};

#pragma unroll 2
    for (int c = 0; c < KCHUNKS; ++c) {
      v8h s0 = *(const v8h*)(aBase + c * 32);
      v8h s1 = *(const v8h*)(aBase + c * 32 + 8);
      v16h sv = __builtin_shufflevector(s0, s1, 0, 1, 2, 3, 4, 5, 6, 7,
                                        8, 9, 10, 11, 12, 13, 14, 15);
      v16h bh = fH[c * 32];
      v16h bl = fL[c * 32];
#pragma unroll
      for (int j = 0; j < SBLOCK; ++j) {
        // A = max(0, 1 - (s - step)^2): exactly 1 iff s == step (integer steps)
        v16h d = sv - stepv[j];
        v16h a = __builtin_elementwise_max(one16 - d * d, zero16);
        // Error-compensated split GEMM: A@(Whi+Wlo), f32 accumulate
        acc[j] = __builtin_amdgcn_wmma_f32_16x16x32_f16(false, a, false, bh,
                                                        (short)0, acc[j], false, false);
        acc[j] = __builtin_amdgcn_wmma_f32_16x16x32_f16(false, a, false, bl,
                                                        (short)0, acc[j], false, false);
      }
    }

    // ---- LIF updates for the 4 blocked steps (reference order per step) ----
#pragma unroll
    for (int j = 0; j < SBLOCK; ++j) {
      v8f Vn = V + 0.025f * (I - V);
      I = 0.9f * I + acc[j];
      float fs = (float)(base + j);
#pragma unroll
      for (int e = 0; e < 8; ++e) {
        if (Vn[e] > 1.0f) {
          if (first[e] == 0.0f) first[e] = fs;  // earliest spiking step
          Vn[e] = 0.0f;                          // reset
        }
      }
      V = Vn;
    }
  }

  // ---- store: C layout -> element e is row M = e + (lane>=16 ? 8 : 0), col N = lane&15
  const int nCol  = ot * 16 + (lane & 15);
  const int mBase = bbase + bs * 16 + ((lane >> 4) ? 8 : 0);
#pragma unroll
  for (int e = 0; e < 8; ++e) {
    float v = first[e];
    out[(size_t)(mBase + e) * OUT_DIM + nCol] = (v == 0.0f) ? 20.0f : v;
  }
}

extern "C" void kernel_launch(void* const* d_in, const int* in_sizes, int n_in,
                              void* d_out, int out_size, void* d_ws, size_t ws_size,
                              hipStream_t stream) {
  (void)in_sizes; (void)n_in; (void)out_size; (void)ws_size;
  const float* inp = (const float*)d_in[0];   // [256, 2048] f32 spike times
  const float* W   = (const float*)d_in[1];   // [2048, 2048] f32 weights
  float* out = (float*)d_out;                 // [256, 2048] f32 first-spike step

  // ws layout: [Whi frags 8MB][Wlo frags 8MB][spike steps f16 1MB] = 17MB
  _Float16* hi = (_Float16*)d_ws;
  _Float16* lo = hi + FRAG_HALVES;
  _Float16* sh = lo + FRAG_HALVES;

  (void)hipFuncSetAttribute(reinterpret_cast<const void*>(snn_ttfs_kernel),
                            hipFuncAttributeMaxDynamicSharedMemorySize, LDS_BYTES);

  prep_w_kernel<<<dim3((OUT_TILES * KCHUNKS * 32) / 256), 256, 0, stream>>>(W, hi, lo);
  prep_s_kernel<<<dim3((BATCH * IN_DIM) / 256), 256, 0, stream>>>(inp, sh);

  dim3 grid(OUT_DIM / 32, BATCH / WG_BATCH);  // 64 x 4 = 256 workgroups
  snn_ttfs_kernel<<<grid, 256, LDS_BYTES, stream>>>(hi, lo, sh, out);
}
